// shPLRNN_88124138979738
// MI455X (gfx1250) — matrix operand order repaired and checked by reading, not compiled, not run on hardware
//
#include <hip/hip_runtime.h>

// shPLRNN on MI455X (gfx1250):
//  S=1024 seq steps, B=64 batch, I=128, H=512, SH=1024.
//  - prep kernel: f32 -> f16 conversion of input/W1/W2/C_w/hx, reset grid barrier.
//  - persistent scan kernel: 16 WGs x 256 threads (8 wave32), grid barrier via
//    agent-scope atomics, all GEMM work on v_wmma_f32_16x16x32_f16 with f32
//    accumulators. Weights (2.25MB f16) stay L2-resident (192MB L2).
//  Phase A uses 1x2 (mt x nt) register blocking per wave: A fragment loaded
//  once, two independent accumulator chains -> 3 b128 loads per WMMA instead
//  of 4, and two dependency-free WMMA streams for the scheduler.

#define S_LEN 1024
#define BATCH 64
#define IDIM 128
#define HDIM 512
#define SHDIM 1024
#define NWG 16
#define WAVES_PER_WG 8
#define NWAVES (NWG * WAVES_PER_WG) // 128

typedef __attribute__((ext_vector_type(16))) _Float16 v16h;
typedef __attribute__((ext_vector_type(8)))  _Float16 v8h;
typedef __attribute__((ext_vector_type(8)))  float    v8f;

// ---------------- workspace layout (bytes) ----------------
// all offsets multiples of 256
#define WS_CNT     0                                   // 256 B (barrier counter)
#define WS_W1H     256                                 // H*SH*2   = 1 MiB
#define WS_W2H     (WS_W1H + HDIM * SHDIM * 2)         // SH*H*2   = 1 MiB
#define WS_CWH     (WS_W2H + SHDIM * HDIM * 2)         // H*I*2    = 128 KiB
#define WS_HH      (WS_CWH + HDIM * IDIM * 2)          // B*H*2    = 64 KiB
#define WS_RELU    (WS_HH  + BATCH * HDIM * 2)         // B*SH*2   = 128 KiB
#define WS_IN16    (WS_RELU + BATCH * SHDIM * 2)       // S*B*I*2  = 16 MiB
// total ~ 18.3 MiB

// ---------------- WMMA fragment helpers ----------------
// A operand (16x32 f16, M x K), source row-major [M][K] with leading dim lda.
// lanes 0-15: VGPR0-3 = K 0..7,   VGPR4-7 = K 8..15   (M = lane)
// lanes16-31: VGPR0-3 = K 16..23, VGPR4-7 = K 24..31  (M = lane-16)
__device__ __forceinline__ v16h load_frag_a(const _Float16* __restrict__ base,
                                            int lda, int lane, int kbase) {
  const _Float16* p = base + (size_t)(lane & 15) * lda + kbase + ((lane >> 4) << 4);
  union { v16h v; v8h h[2]; } u;
  u.h[0] = *(const v8h*)(p);
  u.h[1] = *(const v8h*)(p + 8);
  return u.v;
}

// B operand (32x16 f16, K x N), source stored [N][K] row-major with leading dim ldb
// (i.e. W such that out = act @ W^T). Per lane: 16 contiguous K values at column N.
// lanes 0-15: K 0..15 (N = lane); lanes 16-31: K 16..31 (N = lane-16).
__device__ __forceinline__ v16h load_frag_b(const _Float16* __restrict__ base,
                                            int ldb, int lane, int kbase) {
  const _Float16* p = base + (size_t)(lane & 15) * ldb + kbase + ((lane >> 4) << 4);
  return *(const v16h*)p; // 32B, 32B-aligned for all our ldb/kbase
}

__device__ __forceinline__ v8f wmma16(v16h a, v16h b, v8f c) {
  return __builtin_amdgcn_wmma_f32_16x16x32_f16(false, a, false, b, (short)0, c,
                                                false, false);
}

// ---------------- grid barrier (monotonic counter) ----------------
__device__ __forceinline__ void grid_sync(unsigned* cnt, unsigned target) {
  __threadfence(); // make this thread's global stores device-visible
  __syncthreads();
  if (threadIdx.x == 0) {
    __hip_atomic_fetch_add(cnt, 1u, __ATOMIC_RELEASE, __HIP_MEMORY_SCOPE_AGENT);
    while (__hip_atomic_load(cnt, __ATOMIC_ACQUIRE, __HIP_MEMORY_SCOPE_AGENT) < target) {
      __builtin_amdgcn_s_sleep(8);
    }
  }
  __syncthreads();
}

// ---------------- prep: f32 -> f16 staging + barrier reset ----------------
__global__ void shplrnn_prep(const float* __restrict__ input,
                             const float* __restrict__ hx,
                             const float* __restrict__ W1,
                             const float* __restrict__ W2,
                             const float* __restrict__ Cw,
                             _Float16* __restrict__ in16,
                             _Float16* __restrict__ W1h,
                             _Float16* __restrict__ W2h,
                             _Float16* __restrict__ Cwh,
                             _Float16* __restrict__ hh,
                             unsigned* __restrict__ cnt) {
  size_t i = (size_t)blockIdx.x * blockDim.x + threadIdx.x;
  size_t stride = (size_t)gridDim.x * blockDim.x;
  if (i == 0) *cnt = 0u;
  const size_t nIn = (size_t)S_LEN * BATCH * IDIM;
  const size_t nW1 = (size_t)HDIM * SHDIM;
  const size_t nW2 = (size_t)SHDIM * HDIM;
  const size_t nCw = (size_t)HDIM * IDIM;
  const size_t nHh = (size_t)BATCH * HDIM;
  const size_t total = nIn + nW1 + nW2 + nCw + nHh;
  for (size_t j = i; j < total; j += stride) {
    size_t k = j;
    if (k < nIn) { in16[k] = (_Float16)input[k]; continue; }
    k -= nIn;
    if (k < nW1) { W1h[k] = (_Float16)W1[k]; continue; }
    k -= nW1;
    if (k < nW2) { W2h[k] = (_Float16)W2[k]; continue; }
    k -= nW2;
    if (k < nCw) { Cwh[k] = (_Float16)Cw[k]; continue; }
    k -= nCw;
    hh[k] = (_Float16)hx[k];
  }
}

// ---------------- persistent scan kernel ----------------
__global__ __launch_bounds__(256, 1) void shplrnn_scan(
    const float* __restrict__ hx,        // [B,H]
    const float* __restrict__ Avec,      // [H]
    const float* __restrict__ b1,        // [H]
    const float* __restrict__ b2,        // [SH]
    const float* __restrict__ Cb,        // [H]
    const _Float16* __restrict__ in16,   // [S,B,I]
    const _Float16* __restrict__ W1h,    // [H,SH]  (N=H, K=SH)
    const _Float16* __restrict__ W2h,    // [SH,H]  (N=SH, K=H)
    const _Float16* __restrict__ Cwh,    // [H,I]   (N=H, K=I)
    _Float16* __restrict__ hh,           // [B,H]  f16 copy of current h
    _Float16* __restrict__ relu16,       // [B,SH] f16 relu_term
    float* __restrict__ out,             // [S,B,H] outputs, then [B,H] h_last
    unsigned* __restrict__ cnt) {
  const int lane  = threadIdx.x & 31;
  const int wave  = threadIdx.x >> 5;
  const int gwave = blockIdx.x * WAVES_PER_WG + wave; // 0..127
  const int nloc  = lane & 15;
  const int r0    = (lane >> 4) << 3; // accumulator row base: 0 or 8

  // Phase A blocking: 1 mt x 2 nt per wave (A frag shared, 2 acc chains).
  const int a_mt  = gwave & 3;        // 4 row tiles (B=64)
  const int a_nt0 = (gwave >> 2) * 2; // 64 col tiles (SH=1024), two per wave
  // Phase B: one 16x16 tile per wave.
  const int b_mt = gwave & 3;   // 4 row tiles
  const int b_nt = gwave >> 2;  // 32 col tiles (H=512)

  unsigned round = 0;

  for (int t = 0; t < S_LEN; ++t) {
    // ===== Phase A: relu16 = relu(hh @ W2h^T + b2)  (64 x 1024) =====
    {
      v8f acc0 = {};
      v8f acc1 = {};
      const _Float16* Abase = hh  + (size_t)(a_mt * 16) * HDIM;
      const _Float16* B0    = W2h + (size_t)(a_nt0 * 16) * HDIM;
      const _Float16* B1    = B0 + (size_t)16 * HDIM;
      #pragma unroll 4
      for (int kb = 0; kb < HDIM; kb += 32) {
        v16h a  = load_frag_a(Abase, HDIM, lane, kb);
        v16h b0 = load_frag_b(B0, HDIM, lane, kb);
        v16h b1f = load_frag_b(B1, HDIM, lane, kb);
        acc0 = wmma16(a, b0, acc0);
        acc1 = wmma16(a, b1f, acc1);
      }
      #pragma unroll
      for (int ti = 0; ti < 2; ++ti) {
        const v8f acc = ti ? acc1 : acc0;
        const int n = (a_nt0 + ti) * 16 + nloc;
        const float b2v = b2[n];
        #pragma unroll
        for (int r = 0; r < 8; ++r) {
          const int m = a_mt * 16 + r0 + r;
          float x = acc[r] + b2v;
          relu16[(size_t)m * SHDIM + n] = (_Float16)fmaxf(x, 0.0f);
        }
      }
    }
    ++round;
    grid_sync(cnt, round * NWG);

    // ===== Phase B: h_new = A*h + relu16 @ W1h^T + b1 + (in16[t] @ Cwh^T + Cb) =====
    {
      v8f acc = {};

      // fused cx: input16[t] (64x128) @ C_w^T
      const _Float16* Ai = in16 + (size_t)t * BATCH * IDIM + (size_t)(b_mt * 16) * IDIM;
      const _Float16* Bc = Cwh + (size_t)(b_nt * 16) * IDIM;
      #pragma unroll
      for (int kb = 0; kb < IDIM; kb += 32) {
        v16h a = load_frag_a(Ai, IDIM, lane, kb);
        v16h b = load_frag_b(Bc, IDIM, lane, kb);
        acc = wmma16(a, b, acc);
      }

      // relu16 (64x1024) @ W1^T
      const _Float16* Ar = relu16 + (size_t)(b_mt * 16) * SHDIM;
      const _Float16* Bw = W1h + (size_t)(b_nt * 16) * SHDIM;
      #pragma unroll 4
      for (int kb = 0; kb < SHDIM; kb += 32) {
        v16h a = load_frag_a(Ar, SHDIM, lane, kb);
        v16h b = load_frag_b(Bw, SHDIM, lane, kb);
        acc = wmma16(a, b, acc);
      }

      const int n = b_nt * 16 + nloc;
      const float Av  = Avec[n];
      const float b1v = b1[n];
      const float cbv = Cb[n];
      const float* hp = (t == 0) ? hx : (out + (size_t)(t - 1) * BATCH * HDIM);
      float* op = out + (size_t)t * BATCH * HDIM;
      #pragma unroll
      for (int r = 0; r < 8; ++r) {
        const int m = b_mt * 16 + r0 + r;
        const float hn = Av * hp[(size_t)m * HDIM + n] + acc[r] + b1v + cbv;
        op[(size_t)m * HDIM + n] = hn;
        hh[(size_t)m * HDIM + n] = (_Float16)hn;
        if (t == S_LEN - 1) {
          out[(size_t)S_LEN * BATCH * HDIM + (size_t)m * HDIM + n] = hn;
        }
      }
    }
    ++round;
    grid_sync(cnt, round * NWG);
  }
}

extern "C" void kernel_launch(void* const* d_in, const int* in_sizes, int n_in,
                              void* d_out, int out_size, void* d_ws, size_t ws_size,
                              hipStream_t stream) {
  // setup_inputs order:
  // 0: input (S,B,I) 1: hx (1,B,H) 2: A (H) 3: W1 (H,SH) 4: W2 (SH,H)
  // 5: b1 (H) 6: b2 (SH) 7: C_w (H,I) 8: C_b (H)
  const float* input = (const float*)d_in[0];
  const float* hx    = (const float*)d_in[1];
  const float* Avec  = (const float*)d_in[2];
  const float* W1    = (const float*)d_in[3];
  const float* W2    = (const float*)d_in[4];
  const float* b1    = (const float*)d_in[5];
  const float* b2    = (const float*)d_in[6];
  const float* Cw    = (const float*)d_in[7];
  const float* Cb    = (const float*)d_in[8];
  float* out = (float*)d_out;

  char* ws = (char*)d_ws;
  unsigned*  cnt    = (unsigned*)(ws + WS_CNT);
  _Float16*  W1h    = (_Float16*)(ws + WS_W1H);
  _Float16*  W2h    = (_Float16*)(ws + WS_W2H);
  _Float16*  Cwh    = (_Float16*)(ws + WS_CWH);
  _Float16*  hh     = (_Float16*)(ws + WS_HH);
  _Float16*  relu16 = (_Float16*)(ws + WS_RELU);
  _Float16*  in16   = (_Float16*)(ws + WS_IN16);

  // Stage f16 operands + reset barrier (runs every call -> deterministic).
  shplrnn_prep<<<4096, 256, 0, stream>>>(input, hx, W1, W2, Cw,
                                         in16, W1h, W2h, Cwh, hh, cnt);

  // Persistent scan: 16 WGs x 8 wave32.
  shplrnn_scan<<<NWG, 256, 0, stream>>>(hx, Avec, b1, b2, Cb,
                                        in16, W1h, W2h, Cwh,
                                        hh, relu16, out, cnt);
}
